// MPAdaptive_54039278518613
// MI455X (gfx1250) — matrix operand order repaired and checked by reading, not compiled, not run on hardware
//
#include <hip/hip_runtime.h>

// ---------------------------------------------------------------------------
// MI455X (gfx1250) fused message-passing (attention-style) kernel.
// B=256, N=512, H=64, F1=8. All heavy contractions via v_wmma_f32_16x16x32_bf16.
// ---------------------------------------------------------------------------

typedef __attribute__((ext_vector_type(16))) __bf16 v16bf;
typedef __attribute__((ext_vector_type(8)))  float  v8f;

#define NN 512
#define HH 64
#define F1 8

// ---- WMMA helpers -----------------------------------------------------------
// Fragment layouts used here (wave32; contiguous k order, not the ISA table order):
//  A (16x32 bf16): lane<16 -> row=lane,   K = k0+[0..15]
//                  lane>=16 -> row=lane-16, K = k0+16+[0..15]
//  B (32x16 bf16): lane<16 -> col=lane,    K = k0+[0..15]
//                  lane>=16 -> col=lane-16, K = k0+16+[0..15]
//  C/D (16x16 f32): elem v -> row = v + 8*(lane>=16), col = lane&15

__device__ inline v16bf load_a_frag(const __bf16* src, int stride, int k0) {
    int lane = threadIdx.x & 31;
    int row  = lane & 15;
    int kb   = k0 + ((lane < 16) ? 0 : 16);
    const __bf16* p = src + row * stride + kb;
    v16bf a;
#pragma unroll
    for (int t = 0; t < 8; ++t) { a[t] = p[t]; a[8 + t] = p[8 + t]; }
    return a;
}

// srcT is K-major per column: srcT[n][k] contiguous in k.
__device__ inline v16bf load_b_frag(const __bf16* srcT, int stride, int n0, int k0) {
    int lane = threadIdx.x & 31;
    int n    = n0 + (lane & 15);
    int kb   = k0 + ((lane < 16) ? 0 : 16);
    const __bf16* p = srcT + n * stride + kb;
    v16bf bfr;
#pragma unroll
    for (int t = 0; t < 16; ++t) bfr[t] = p[t];
    return bfr;
}

__device__ inline v8f wmma_bf16(v16bf a, v16bf b, v8f c) {
    return __builtin_amdgcn_wmma_f32_16x16x32_bf16(false, a, false, b,
                                                   (short)0, c, false, false);
}

// ---------------------------------------------------------------------------
// Kernel 0: weight conversion.
//   WadjT[n][k] = W_adj[k][n]  (bf16, 64x64)   -> B-matrix for q = h @ W_adj
//   WmsgT[n][k] = W_msg[k][n]  (bf16, 64x64)   -> B-matrix for m = h @ W_msg
//   Wih_pad[n][k] = W_ih[n][k] (bf16, 192x96, k>=72 zero) -> B for x @ W_ih^T
//   Whh_b[n][k]   = W_hh[n][k] (bf16, 192x64)             -> B for h @ W_hh^T
// ---------------------------------------------------------------------------
__global__ __launch_bounds__(256) void prep_weights(
    const float* __restrict__ W_adj, const float* __restrict__ W_msg,
    const float* __restrict__ W_ih,  const float* __restrict__ W_hh,
    __bf16* __restrict__ WadjT, __bf16* __restrict__ WmsgT,
    __bf16* __restrict__ Wih_pad, __bf16* __restrict__ Whh_b) {
    int idx = blockIdx.x * blockDim.x + threadIdx.x;
    if (idx < 64 * 64) {
        int n = idx >> 6, k = idx & 63;
        WadjT[n * 64 + k] = (__bf16)W_adj[k * 64 + n];
        WmsgT[n * 64 + k] = (__bf16)W_msg[k * 64 + n];
    }
    if (idx < 192 * 96) {
        int n = idx / 96, k = idx % 96;
        Wih_pad[idx] = (k < 72) ? (__bf16)W_ih[n * 72 + k] : (__bf16)0.0f;
    }
    if (idx < 192 * 64) {
        Whh_b[idx] = (__bf16)W_hh[idx];
    }
}

// ---------------------------------------------------------------------------
// Kernel 1: per (b, 16-row tile):
//   h_b  = bf16(h)                       [B,N,H] row-major (K-matrix for S)
//   m_T  = bf16(h @ W_msg + b_msg)^T     [B,H,N] (B-matrix for A @ m)
// ---------------------------------------------------------------------------
__global__ __launch_bounds__(256) void prep_hm(
    const float* __restrict__ h, const float* __restrict__ b_msg,
    const __bf16* __restrict__ WmsgT,
    __bf16* __restrict__ h_b, __bf16* __restrict__ m_T) {
    int b  = blockIdx.y;
    int j0 = blockIdx.x * 16;
    int tid = threadIdx.x, wave = tid >> 5, lane = tid & 31;

    __shared__ __bf16 s_hb[16 * HH];

    size_t base = ((size_t)b * NN + j0) * HH;
    for (int idx = tid; idx < 16 * HH; idx += 256) {
        float f = h[base + idx];
        __bf16 v = (__bf16)f;
        s_hb[idx] = v;
        h_b[base + idx] = v;
    }
    __syncthreads();

    if (wave < 4) {
        int nt = wave;
        v8f a0 = {}, a1 = {};
        {
            v16bf x0 = load_a_frag(s_hb, HH, 0);
            v16bf w0 = load_b_frag(WmsgT, HH, nt * 16, 0);
            v16bf x1 = load_a_frag(s_hb, HH, 32);
            v16bf w1 = load_b_frag(WmsgT, HH, nt * 16, 32);
            a0 = wmma_bf16(x0, w0, a0);
            a1 = wmma_bf16(x1, w1, a1);
        }
        int col = nt * 16 + (lane & 15);
        float bias = b_msg[col];
#pragma unroll
        for (int v = 0; v < 8; ++v) {
            int row = v + 8 * (lane >> 4);
            m_T[((size_t)b * HH + col) * NN + j0 + row] =
                (__bf16)(a0[v] + a1[v] + bias);
        }
    }
}

// ---------------------------------------------------------------------------
// Kernel 2: fused attention + GRU. One block = (batch b, 16 rows of N).
// ---------------------------------------------------------------------------
__global__ __launch_bounds__(256) void mp_fused(
    const float* __restrict__ h,    const float* __restrict__ jets,
    const float* __restrict__ mask,
    const float* __restrict__ b_ih, const float* __restrict__ b_hh,
    const __bf16* __restrict__ h_b, const __bf16* __restrict__ m_T,
    const __bf16* __restrict__ WadjT,
    const __bf16* __restrict__ Wih_pad, const __bf16* __restrict__ Whh_b,
    float* __restrict__ out) {
    int b  = blockIdx.y;
    int i0 = blockIdx.x * 16;
    int tid = threadIdx.x, wave = tid >> 5, lane = tid & 31;

    __shared__ float  s_S[16 * NN];      // 32 KB  score strip (fp32)
    __shared__ __bf16 s_A[16 * NN];      // 16 KB  attention probs (bf16)
    __shared__ __bf16 s_q[16 * HH];      //  2 KB  q = h_tile @ W_adj
    __shared__ __bf16 s_hb[16 * HH];     //  2 KB  h tile bf16
    __shared__ float  s_hf[16 * HH];     //  4 KB  h tile fp32 (GRU final mix)
    __shared__ __bf16 s_x1[16 * 96];     //  3 KB  [message | jets | pad]
    __shared__ float  s_g1[16 * 192];    // 12 KB  gi
    __shared__ float  s_g2[16 * 192];    // 12 KB  gh
    __shared__ float  s_p[2][1024];      //  8 KB  K-split partials of A@m
    __shared__ float  s_red[16 * 16];    //  1 KB  softmax partials
    __shared__ float  s_stat[16];        //        row max / row sum

    // ---- Phase 0a: stage h tile (fp32 + bf16) and jets into x1 -------------
    size_t hbase = ((size_t)b * NN + i0) * HH;
    for (int idx = tid; idx < 16 * HH; idx += 256) {
        float f = h[hbase + idx];
        s_hf[idx] = f;
        s_hb[idx] = (__bf16)f;
    }
    for (int idx = tid; idx < 16 * 32; idx += 256) {   // x1 cols 64..95
        int r = idx >> 5, c = idx & 31;
        __bf16 v = (__bf16)0.0f;
        if (c < F1) v = (__bf16)jets[((size_t)b * NN + i0 + r) * F1 + c];
        s_x1[r * 96 + 64 + c] = v;
    }
    __syncthreads();

    // ---- Phase 0b: q = h_tile @ W_adj (waves 0..3, one 16-col tile each) ---
    if (wave < 4) {
        int nt = wave;
        v8f a0 = {}, a1 = {};
        {
            v16bf x0 = load_a_frag(s_hb, HH, 0);
            v16bf w0 = load_b_frag(WadjT, HH, nt * 16, 0);
            v16bf x1 = load_a_frag(s_hb, HH, 32);
            v16bf w1 = load_b_frag(WadjT, HH, nt * 16, 32);
            a0 = wmma_bf16(x0, w0, a0);
            a1 = wmma_bf16(x1, w1, a1);
        }
        int col = nt * 16 + (lane & 15);
#pragma unroll
        for (int v = 0; v < 8; ++v)
            s_q[(v + 8 * (lane >> 4)) * HH + col] = (__bf16)(a0[v] + a1[v]);
    }
    __syncthreads();

    // ---- Phase 1: S = q @ h^T, column mask. 32 j-tiles over 8 waves --------
    // Also prefetch this batch's m_T strip (needed in phase 3) into cache.
    {
        const char* mpf = (const char*)(m_T + (size_t)b * HH * NN);
        for (int off = tid * 256; off < HH * NN * 2; off += 256 * 256)
            __builtin_prefetch(mpf + off, 0, 1);

        v16bf aq0 = load_a_frag(s_q, HH, 0);
        v16bf aq1 = load_a_frag(s_q, HH, 32);
        for (int jt = wave; jt < 32; jt += 8) {
            v8f c0 = {}, c1 = {};
            const __bf16* hrow = h_b + ((size_t)b * NN + jt * 16) * HH;
            v16bf b0 = load_b_frag(hrow, HH, 0, 0);
            v16bf b1 = load_b_frag(hrow, HH, 0, 32);
            c0 = wmma_bf16(aq0, b0, c0);
            c1 = wmma_bf16(aq1, b1, c1);
            int col = jt * 16 + (lane & 15);
            float mj = mask[(size_t)b * NN + col];
#pragma unroll
            for (int v = 0; v < 8; ++v) {
                int row = v + 8 * (lane >> 4);
                s_S[row * NN + col] = (mj > 0.0f) ? (c0[v] + c1[v]) : -3.4e38f;
            }
        }
    }
    __syncthreads();

    // ---- Phase 2: masked softmax per row, write bf16 probs -----------------
    {
        int r = tid >> 4, sub = tid & 15;
        float pmax = -3.4e38f;
        for (int c = sub; c < NN; c += 16) pmax = fmaxf(pmax, s_S[r * NN + c]);
        s_red[r * 16 + sub] = pmax;
        __syncthreads();
        if (sub == 0) {
            float m = s_red[r * 16];
#pragma unroll
            for (int t = 1; t < 16; ++t) m = fmaxf(m, s_red[r * 16 + t]);
            s_stat[r] = m;
        }
        __syncthreads();
        float rmax = s_stat[r];
        float psum = 0.0f;
        for (int c = sub; c < NN; c += 16) {
            float e = __expf(s_S[r * NN + c] - rmax);
            s_S[r * NN + c] = e;
            psum += e;
        }
        s_red[r * 16 + sub] = psum;
        __syncthreads();
        if (sub == 0) {
            float s = 0.0f;
#pragma unroll
            for (int t = 0; t < 16; ++t) s += s_red[r * 16 + t];
            s_stat[r] = s + 1e-30f;
        }
        __syncthreads();
        float scale = mask[(size_t)b * NN + i0 + r] / s_stat[r];
        for (int c = sub; c < NN; c += 16)
            s_A[r * NN + c] = (__bf16)(s_S[r * NN + c] * scale);
    }
    __syncthreads();

    // ---- Phase 3: message = tanh(A @ m). 4 n-tiles x 2 K-halves / 8 waves --
    // Two independent accumulator chains to hide the WMMA->WMMA RAW hazard.
    {
        int nt = wave & 3, kh = wave >> 2;
        v8f acc0 = {}, acc1 = {};
        const __bf16* mrow = m_T + ((size_t)b * HH + nt * 16) * NN;
        for (int kc = kh * 256; kc < kh * 256 + 256; kc += 64) {
            v16bf a0 = load_a_frag(s_A, NN, kc);
            v16bf bb0 = load_b_frag(mrow, NN, 0, kc);
            v16bf a1 = load_a_frag(s_A, NN, kc + 32);
            v16bf bb1 = load_b_frag(mrow, NN, 0, kc + 32);
            acc0 = wmma_bf16(a0, bb0, acc0);
            acc1 = wmma_bf16(a1, bb1, acc1);
        }
        float* pp = &s_p[kh][nt * 256 + lane * 8];
#pragma unroll
        for (int v = 0; v < 8; ++v) pp[v] = acc0[v] + acc1[v];
    }
    __syncthreads();
    for (int idx = tid; idx < 1024; idx += 256) {
        int nt = idx >> 8, ln = (idx >> 3) & 31, v = idx & 7;
        int row = v + 8 * (ln >> 4), col = nt * 16 + (ln & 15);
        float msg = tanhf(s_p[0][idx] + s_p[1][idx]);
        s_x1[row * 96 + col] = (__bf16)msg;
    }
    __syncthreads();

    // ---- Phase 4: GRU gate GEMMs: g1 = x1 @ W_ih^T, g2 = h @ W_hh^T --------
    for (int nt = wave; nt < 12; nt += 8) {
        v8f acc = {};
#pragma unroll
        for (int kc = 0; kc < 96; kc += 32) {
            v16bf a  = load_a_frag(s_x1, 96, kc);
            v16bf bb = load_b_frag(Wih_pad, 96, nt * 16, kc);
            acc = wmma_bf16(a, bb, acc);
        }
        int col = nt * 16 + (lane & 15);
#pragma unroll
        for (int v = 0; v < 8; ++v)
            s_g1[(v + 8 * (lane >> 4)) * 192 + col] = acc[v];
    }
    for (int nt = wave; nt < 12; nt += 8) {
        v8f a0 = {}, a1 = {};
        {
            v16bf x0 = load_a_frag(s_hb, HH, 0);
            v16bf w0 = load_b_frag(Whh_b, HH, nt * 16, 0);
            v16bf x1 = load_a_frag(s_hb, HH, 32);
            v16bf w1 = load_b_frag(Whh_b, HH, nt * 16, 32);
            a0 = wmma_bf16(x0, w0, a0);
            a1 = wmma_bf16(x1, w1, a1);
        }
        int col = nt * 16 + (lane & 15);
#pragma unroll
        for (int v = 0; v < 8; ++v)
            s_g2[(v + 8 * (lane >> 4)) * 192 + col] = a0[v] + a1[v];
    }
    __syncthreads();

    // ---- Phase 5: GRU pointwise + output -----------------------------------
    for (int idx = tid; idx < 16 * HH; idx += 256) {
        int r = idx >> 6, c = idx & 63;
        float ir = s_g1[r * 192 + c]        + b_ih[c];
        float iz = s_g1[r * 192 + 64 + c]   + b_ih[64 + c];
        float ic = s_g1[r * 192 + 128 + c]  + b_ih[128 + c];
        float hr = s_g2[r * 192 + c]        + b_hh[c];
        float hz = s_g2[r * 192 + 64 + c]   + b_hh[64 + c];
        float hc = s_g2[r * 192 + 128 + c]  + b_hh[128 + c];
        float rg = 1.0f / (1.0f + __expf(-(ir + hr)));
        float z  = 1.0f / (1.0f + __expf(-(iz + hz)));
        float nn = tanhf(ic + rg * hc);
        out[hbase + idx] = (1.0f - z) * nn + z * s_hf[idx];
    }
}

// ---------------------------------------------------------------------------
extern "C" void kernel_launch(void* const* d_in, const int* in_sizes, int n_in,
                              void* d_out, int out_size, void* d_ws, size_t ws_size,
                              hipStream_t stream) {
    const float* h     = (const float*)d_in[0];
    const float* jets  = (const float*)d_in[1];
    const float* mask  = (const float*)d_in[2];
    const float* W_adj = (const float*)d_in[3];
    const float* W_msg = (const float*)d_in[4];
    const float* b_msg = (const float*)d_in[5];
    const float* W_ih  = (const float*)d_in[6];
    const float* b_ih  = (const float*)d_in[7];
    const float* W_hh  = (const float*)d_in[8];
    const float* b_hh  = (const float*)d_in[9];
    float* out = (float*)d_out;

    const int B = in_sizes[0] / (NN * HH);

    // Workspace carve-up (256B aligned)
    size_t off = 0;
    auto carve = [&](size_t bytes) {
        void* p = (char*)d_ws + off;
        off += (bytes + 255) & ~(size_t)255;
        return p;
    };
    __bf16* h_b     = (__bf16*)carve((size_t)B * NN * HH * sizeof(__bf16));
    __bf16* m_T     = (__bf16*)carve((size_t)B * HH * NN * sizeof(__bf16));
    __bf16* WadjT   = (__bf16*)carve(64 * 64 * sizeof(__bf16));
    __bf16* WmsgT   = (__bf16*)carve(64 * 64 * sizeof(__bf16));
    __bf16* Wih_pad = (__bf16*)carve(192 * 96 * sizeof(__bf16));
    __bf16* Whh_b   = (__bf16*)carve(192 * 64 * sizeof(__bf16));
    (void)ws_size; (void)n_in; (void)out_size;

    prep_weights<<<(192 * 96 + 255) / 256, 256, 0, stream>>>(
        W_adj, W_msg, W_ih, W_hh, WadjT, WmsgT, Wih_pad, Whh_b);

    prep_hm<<<dim3(NN / 16, B), 256, 0, stream>>>(h, b_msg, WmsgT, h_b, m_T);

    mp_fused<<<dim3(NN / 16, B), 256, 0, stream>>>(
        h, jets, mask, b_ih, b_hh, h_b, m_T, WadjT, Wih_pad, Whh_b, out);
}
